// Model_75840532513009
// MI455X (gfx1250) — matrix operand order repaired and checked by reading, not compiled, not run on hardware
//
#include <hip/hip_runtime.h>
#include <math.h>

// ---------------------------------------------------------------------------
// Informer forward on MI455X (gfx1250, wave32).
// - Weights pre-converted+transposed once per launch to f16 [N][K].
// - GEMM: v_wmma_f32_16x16x32_f16; A staged fp32->f16 via 128-bit loads +
//   v_cvt_pk_f16_f32; B tile fetched by the Tensor Data Mover
//   (tensor_load_to_lds, TENSORcnt) with descriptor padding recreating the
//   bank-conflict-free 40-half LDS pitch. Wave 0 overlaps TDM with A staging.
// ---------------------------------------------------------------------------

typedef __attribute__((ext_vector_type(16))) _Float16 v16h;
typedef __attribute__((ext_vector_type(8)))  float    v8f;
typedef __attribute__((ext_vector_type(4)))  float    f4v;
typedef __attribute__((ext_vector_type(4)))  _Float16 h4v;
typedef __attribute__((ext_vector_type(8)))  _Float16 h8v;
typedef __attribute__((ext_vector_type(4)))  unsigned u32x4;
typedef __attribute__((ext_vector_type(8)))  int      i32x8;
typedef __attribute__((ext_vector_type(4)))  int      i32x4;

#ifndef __has_builtin
#define __has_builtin(x) 0
#endif
#if __has_builtin(__builtin_amdgcn_tensor_load_to_lds) && \
    __has_builtin(__builtin_amdgcn_s_wait_tensorcnt)
#define HAVE_TDM 1
#endif

#define GEMM_TM 128
#define GEMM_TN 64
#define GEMM_TK 32
#define A_PITCH (GEMM_TK + 8)   // 40 halves = 80 B rows (16B aligned)
#define B_PITCH (GEMM_TK + 8)

// flags: 1 = add bias, 2 = exact GELU, 4 = accumulate into C
__global__ __launch_bounds__(256)
void gemm_wmma_kernel(const float* __restrict__ A, const _Float16* __restrict__ Bt,
                      const float* __restrict__ bias, float* __restrict__ C,
                      int M, int N, int K, int flags)
{
    __shared__ __align__(16) _Float16 As[GEMM_TM][A_PITCH];   // [m][k]
    __shared__ __align__(16) _Float16 Bs[GEMM_TN][B_PITCH];   // [n][k] (pre-transposed)

    const int tid  = threadIdx.x;
    const int wave = tid >> 5;          // 0..7
    const int lane = tid & 31;
    const int wm   = wave >> 1;         // 0..3  (32-row slab)
    const int wn   = wave & 1;          // 0..1  (32-col slab)
    const int rowBase = blockIdx.y * GEMM_TM;
    const int colBase = blockIdx.x * GEMM_TN;

    v8f acc[2][2] = {};

    const int lhalf = lane & 15;
    const int hi    = lane >> 4;        // 0 or 1

    for (int k0 = 0; k0 < K; k0 += GEMM_TK) {
#ifdef HAVE_TDM
        // ---- B tile (64 rows x 32 halves) via Tensor Data Mover ---------
        if (wave == 0) {
            const unsigned ldsb = (unsigned)(size_t)&Bs[0][0];
            const unsigned long long ga =
                (unsigned long long)(size_t)(Bt + (size_t)colBase * K + k0);
            u32x4 g0 = {};
            g0[0] = 1u;                                   // count=1 (valid D#)
            g0[1] = ldsb;                                 // lds_addr
            g0[2] = (unsigned)ga;                         // global_addr[31:0]
            g0[3] = (unsigned)((ga >> 32) & 0x01FFFFFFu)  // global_addr[56:32]
                  | (2u << 30);                           // type=2 (image)
            i32x8 g1 = {};
            g1[0] = (1 << 16)      // data_size = 2 bytes
                  | (1 << 20)      // pad_enable
                  | (3 << 22)      // pad_interval: 16 DWORDs (one 32-half row)
                  | (3 << 25);     // pad_amount: 4 DWORDs (8 halves) -> pitch 40
            g1[1] = (K & 0xFFFF) << 16;                          // tensor_dim0 lo
            g1[2] = ((K >> 16) & 0xFFFF) | ((N & 0xFFFF) << 16); // dim0 hi | dim1 lo
            g1[3] = ((N >> 16) & 0xFFFF) | (GEMM_TK << 16);      // dim1 hi | tile_dim0
            g1[4] = GEMM_TN;                                     // tile_dim1 (tile_dim2=0)
            g1[5] = K;                                           // tensor_dim0_stride lo
            g1[6] = 0; g1[7] = 0;
            i32x4 gz = {};
#if __clang_major__ >= 23
            i32x8 gz8 = {};
            __builtin_amdgcn_tensor_load_to_lds(g0, g1, gz, gz, gz8, 0);
#else
            __builtin_amdgcn_tensor_load_to_lds(g0, g1, gz, gz, 0);
#endif
        }
#else
        // ---- fallback: direct vectorized f16 copy -----------------------
        {
            const int n  = tid >> 2;
            const int kc = (tid & 3) * 8;
            const _Float16* src = Bt + (size_t)(colBase + n) * K + k0 + kc;
            *(h8v*)&Bs[n][kc] = *(const h8v*)src;
        }
#endif
        // ---- stage A tile (128x32): f32 -> f16, 16 elems / thread -------
        {
            const int r  = tid >> 1;
            const int c  = (tid & 1) * 16;
            const int gr = rowBase + r;
            const float* src = A + (size_t)gr * K + k0 + c;
            if (k0 + GEMM_TK < K) __builtin_prefetch(src + GEMM_TK, 0, 1);
            h8v* dst = (h8v*)&As[r][c];
            if (gr < M) {
                f4v f0 = *(const f4v*)(src + 0);
                f4v f1 = *(const f4v*)(src + 4);
                f4v f2 = *(const f4v*)(src + 8);
                f4v f3 = *(const f4v*)(src + 12);
                h4v c0 = __builtin_convertvector(f0, h4v);
                h4v c1 = __builtin_convertvector(f1, h4v);
                h4v c2 = __builtin_convertvector(f2, h4v);
                h4v c3 = __builtin_convertvector(f3, h4v);
                dst[0] = __builtin_shufflevector(c0, c1, 0,1,2,3,4,5,6,7);
                dst[1] = __builtin_shufflevector(c2, c3, 0,1,2,3,4,5,6,7);
            } else {
                dst[0] = (h8v)0;
                dst[1] = (h8v)0;
            }
        }
#ifdef HAVE_TDM
        if (wave == 0) __builtin_amdgcn_s_wait_tensorcnt((short)0);
#endif
        __syncthreads();

        // ---- fragments: two aligned 16B LDS vector loads per sub-tile ---
        v16h afrag[2], bfrag[2];
#pragma unroll
        for (int s = 0; s < 2; ++s) {
            const int m = wm * 32 + s * 16 + lhalf;
            // A 16x32 layout: half h -> K = (h<8 ? h : h+8) + hi*8
            h8v alo = *(const h8v*)&As[m][hi * 8];
            h8v ahi = *(const h8v*)&As[m][16 + hi * 8];
            afrag[s] = __builtin_shufflevector(alo, ahi,
                0,1,2,3,4,5,6,7,8,9,10,11,12,13,14,15);
            const int n = wn * 32 + s * 16 + lhalf;
            // B 32x16 layout: half h -> K = (h<8 ? h : h+8) + hi*8
            h8v blo = *(const h8v*)&Bs[n][hi * 8];
            h8v bhi = *(const h8v*)&Bs[n][16 + hi * 8];
            bfrag[s] = __builtin_shufflevector(blo, bhi,
                0,1,2,3,4,5,6,7,8,9,10,11,12,13,14,15);
        }

#pragma unroll
        for (int i = 0; i < 2; ++i)
#pragma unroll
            for (int j = 0; j < 2; ++j)
                acc[i][j] = __builtin_amdgcn_wmma_f32_16x16x32_f16(
                    false, afrag[i], false, bfrag[j],
                    (short)0, acc[i][j], false, false);

        __syncthreads();
    }

    // ---- epilogue: bias / GELU / accumulate, per documented C layout ----
#pragma unroll
    for (int i = 0; i < 2; ++i) {
#pragma unroll
        for (int j = 0; j < 2; ++j) {
            const int n = colBase + wn * 32 + j * 16 + lhalf;
            const float bv = (flags & 1) ? bias[n] : 0.0f;
#pragma unroll
            for (int r = 0; r < 8; ++r) {
                const int m = rowBase + wm * 32 + i * 16 + hi * 8 + r;
                if (m < M && n < N) {
                    float v = acc[i][j][r] + bv;
                    if (flags & 2)
                        v = 0.5f * v * (1.0f + erff(v * 0.70710678f));
                    const size_t cidx = (size_t)m * N + n;
                    if (flags & 4) v += C[cidx];
                    C[cidx] = v;
                }
            }
        }
    }
}

// Wt[n*K + k] = (f16) W[k*N + n]   (weights are static: done once per launch)
__global__ void convert_transpose_kernel(const float* __restrict__ W,
                                         _Float16* __restrict__ Wt, int K, int N)
{
    const int idx = blockIdx.x * blockDim.x + threadIdx.x;
    if (idx >= K * N) return;
    const int k = idx % K;
    const int n = idx / K;
    Wt[(size_t)n * K + k] = (_Float16)W[(size_t)k * N + n];
}

// ---- fused online-softmax attention: 8 wave32 queries per block ----------
__global__ __launch_bounds__(256)
void attention_kernel(const float* __restrict__ Q, const float* __restrict__ Kt,
                      const float* __restrict__ V, float* __restrict__ O,
                      int Lq, int Lk, int dmodel, int dh, int causal)
{
    const int q = blockIdx.x * 8 + (threadIdx.x >> 5);
    const int h = blockIdx.y, b = blockIdx.z;
    const int lane = threadIdx.x & 31;
    if (q >= Lq) return;
    const float* qrow = Q + ((size_t)(b * Lq + q) * dmodel) + h * dh;
    const float* kb   = Kt + ((size_t)b * Lk) * dmodel + h * dh;
    const float* vb   = V  + ((size_t)b * Lk) * dmodel + h * dh;
    const int d0 = lane * 2;
    const float q0 = qrow[d0], q1 = qrow[d0 + 1];
    const float scale = rsqrtf((float)dh);

    float m = -3.4e38f, se = 0.0f, a0 = 0.0f, a1 = 0.0f;
    const int kend = causal ? (q + 1) : Lk;
    for (int k = 0; k < kend; ++k) {
        const float* kr = kb + (size_t)k * dmodel;
        float p = q0 * kr[d0] + q1 * kr[d0 + 1];
#pragma unroll
        for (int off = 16; off > 0; off >>= 1)
            p += __shfl_xor(p, off, 32);
        const float s  = p * scale;
        const float mn = fmaxf(m, s);
        const float cf = __expf(m - mn);
        const float e  = __expf(s - mn);
        const float* vr = vb + (size_t)k * dmodel;
        a0 = a0 * cf + e * vr[d0];
        a1 = a1 * cf + e * vr[d0 + 1];
        se = se * cf + e;
        m  = mn;
    }
    const float inv = 1.0f / se;
    float* orow = O + ((size_t)(b * Lq + q) * dmodel) + h * dh;
    orow[d0]     = a0 * inv;
    orow[d0 + 1] = a1 * inv;
}

// ---- LayerNorm (optional fused residual), one wave32 per row, D=512 ------
__global__ __launch_bounds__(256)
void ln_kernel(const float* __restrict__ X, const float* __restrict__ R,
               const float* __restrict__ g, const float* __restrict__ beta,
               float* __restrict__ O, int rows, int D)
{
    const int row  = blockIdx.x * 8 + (threadIdx.x >> 5);
    const int lane = threadIdx.x & 31;
    if (row >= rows) return;
    const float* x = X + (size_t)row * D;
    const float* r = R ? R + (size_t)row * D : nullptr;
    const int per = D / 32;               // 16 for D=512
    float vals[16];
    float s = 0.0f;
    for (int i = 0; i < per; ++i) {
        float v = x[lane + i * 32];
        if (r) v += r[lane + i * 32];
        vals[i] = v; s += v;
    }
#pragma unroll
    for (int off = 16; off > 0; off >>= 1) s += __shfl_xor(s, off, 32);
    const float mean = s / (float)D;
    float vs = 0.0f;
    for (int i = 0; i < per; ++i) { const float d = vals[i] - mean; vs += d * d; }
#pragma unroll
    for (int off = 16; off > 0; off >>= 1) vs += __shfl_xor(vs, off, 32);
    const float rstd = rsqrtf(vs / (float)D + 1e-5f);
    float* o = O + (size_t)row * D;
    for (int i = 0; i < per; ++i) {
        const int c = lane + i * 32;
        o[c] = (vals[i] - mean) * rstd * g[c] + beta[c];
    }
}

// ---- token embedding: 3-tap circular conv + positional + time marks ------
__global__ void embed_kernel(const float* __restrict__ xin, const float* __restrict__ mark,
                             const float* __restrict__ convw, const float* __restrict__ tw,
                             const float* __restrict__ tb, float* __restrict__ out,
                             int Bn, int L, int Cin, int D, int MK)
{
    const int idx = blockIdx.x * blockDim.x + threadIdx.x;
    const int total = Bn * L * D;
    if (idx >= total) return;
    const int d = idx % D;
    const int l = (idx / D) % L;
    const int b = idx / (D * L);
    const int lm = (l - 1 + L) % L, lp = (l + 1) % L;
    const float* x0 = xin + ((size_t)b * L + lm) * Cin;   // roll(x,+1)[l] = x[l-1]
    const float* x1 = xin + ((size_t)b * L + l ) * Cin;
    const float* x2 = xin + ((size_t)b * L + lp) * Cin;   // roll(x,-1)[l] = x[l+1]
    float acc = tb[d];
    for (int c = 0; c < Cin; ++c)
        acc += x0[c] * convw[(0 * Cin + c) * D + d]
             + x1[c] * convw[(1 * Cin + c) * D + d]
             + x2[c] * convw[(2 * Cin + c) * D + d];
    const float* mk = mark + ((size_t)b * L + l) * MK;
    for (int t = 0; t < MK; ++t) acc += mk[t] * tw[t * D + d];
    const int dd = d & ~1;
    const float div = __expf(-logf(10000.0f) * (float)dd / (float)D);
    const float ang = (float)l * div;
    acc += (d & 1) ? __cosf(ang) : __sinf(ang);
    out[idx] = acc;
}

// out[b,l,:] = in[b,(l+shift) mod L,:]
__global__ void roll_rows_kernel(const float* __restrict__ in, float* __restrict__ out,
                                 int Bn, int L, int D, int shift)
{
    const int idx = blockIdx.x * blockDim.x + threadIdx.x;
    const int total = Bn * L * D;
    if (idx >= total) return;
    const int d = idx % D;
    const int l = (idx / D) % L;
    const int b = idx / (D * L);
    const int ls = ((l + shift) % L + L) % L;
    out[idx] = in[((size_t)b * L + ls) * D + d];
}

// per-channel mean/var over all rows
__global__ __launch_bounds__(256)
void chan_stats_kernel(const float* __restrict__ X, float* __restrict__ mean,
                       float* __restrict__ var, int rows, int D)
{
    const int c = blockIdx.x;
    float s = 0.0f, s2 = 0.0f;
    for (int r = threadIdx.x; r < rows; r += 256) {
        const float v = X[(size_t)r * D + c];
        s += v; s2 += v * v;
    }
    __shared__ float sh[256], sh2[256];
    sh[threadIdx.x] = s; sh2[threadIdx.x] = s2;
    __syncthreads();
    for (int st = 128; st > 0; st >>= 1) {
        if (threadIdx.x < st) {
            sh[threadIdx.x]  += sh[threadIdx.x + st];
            sh2[threadIdx.x] += sh2[threadIdx.x + st];
        }
        __syncthreads();
    }
    if (threadIdx.x == 0) {
        const float mm = sh[0] / (float)rows;
        mean[c] = mm;
        var[c]  = sh2[0] / (float)rows - mm * mm;
    }
}

__global__ void bn_elu_kernel(float* __restrict__ X, const float* __restrict__ mean,
                              const float* __restrict__ var, const float* __restrict__ g,
                              const float* __restrict__ beta, int total, int D)
{
    const int idx = blockIdx.x * blockDim.x + threadIdx.x;
    if (idx >= total) return;
    const int c = idx % D;
    float v = (X[idx] - mean[c]) * rsqrtf(var[c] + 1e-5f) * g[c] + beta[c];
    X[idx] = v > 0.0f ? v : expm1f(v);
}

// window 3, stride 2, pad 1 along L
__global__ void maxpool_kernel(const float* __restrict__ X, float* __restrict__ O,
                               int Bn, int L, int D)
{
    const int L2 = L / 2;
    const int idx = blockIdx.x * blockDim.x + threadIdx.x;
    const int total = Bn * L2 * D;
    if (idx >= total) return;
    const int d  = idx % D;
    const int l2 = (idx / D) % L2;
    const int b  = idx / (D * L2);
    float m = -3.4e38f;
    for (int t = -1; t <= 1; ++t) {
        const int l = 2 * l2 + t;
        if (l >= 0 && l < L)
            m = fmaxf(m, X[((size_t)b * L + l) * D + d]);
    }
    O[idx] = m;
}

__global__ void proj_kernel(const float* __restrict__ dec, const float* __restrict__ pw,
                            const float* __restrict__ pb, float* __restrict__ out,
                            int Bn, int Ldec, int predLen, int D, int Cout)
{
    const int idx = blockIdx.x * blockDim.x + threadIdx.x;
    const int total = Bn * predLen * Cout;
    if (idx >= total) return;
    const int c = idx % Cout;
    const int l = (idx / Cout) % predLen;
    const int b = idx / (Cout * predLen);
    const int ls = Ldec - predLen + l;
    const float* row = dec + ((size_t)b * Ldec + ls) * D;
    float acc = pb[c];
    for (int d = 0; d < D; ++d) acc += row[d] * pw[d * Cout + c];
    out[idx] = acc;
}

// ---------------------------------------------------------------------------
extern "C" void kernel_launch(void* const* d_in, const int* in_sizes, int n_in,
                              void* d_out, int out_size, void* d_ws, size_t ws_size,
                              hipStream_t stream)
{
    (void)in_sizes; (void)n_in; (void)out_size; (void)ws_size;
#define IN(i) ((const float*)d_in[(i)])
    const float* x_enc      = IN(0);
    const float* x_mark_enc = IN(1);
    const float* x_dec      = IN(2);
    const float* x_mark_dec = IN(3);
    // params flattened in JAX pytree order (dict keys sorted at each level):
    // 4:dec_emb.conv_w 5:dec_emb.tb 6:dec_emb.tw  7:dec.b1 8:dec.b2
    // 9..16: dec.cross {bk,bo,bq,bv,wk,wo,wq,wv}
    // 17:n1b 18:n1g 19:n2b 20:n2g 21:n3b 22:n3g
    // 23..30: dec.self {bk,bo,bq,bv,wk,wo,wq,wv}
    // 31:dec.w1 32:dec.w2  33:dec_norm.b 34:dec_norm.g
    // 35:enc_convs0.b 36:beta 37:g 38:w
    // 39:enc_emb.conv_w 40:enc_emb.tb 41:enc_emb.tw
    // 42..57: enc_layer0 {attn(bk,bo,bq,bv,wk,wo,wq,wv),b1,b2,n1b,n1g,n2b,n2g,w1,w2}
    // 58..73: enc_layer1 same
    // 74:enc_norm.b 75:enc_norm.g 76:proj_b 77:proj_w

    const int Bn = 8, Lenc = 1024, Ldec = 1024, D = 512, FF = 2048, PRED = 512;

    // ---- workspace carve: fp32 activations then f16 weight area ----
    float* ws = (float*)d_ws;
    size_t off = 0;
    auto alloc = [&](size_t n) { float* p = ws + off; off += n; return p; };
    const size_t MD = (size_t)Bn * Lenc * D;        // 8192*512
    float* encx    = alloc(MD);
    float* x2      = alloc(MD);
    float* tmpa    = alloc(MD);
    float* tq      = alloc(MD);
    float* tk      = alloc(MD);
    float* tv      = alloc(MD);
    float* tctx    = alloc(MD);
    float* ffh     = alloc((size_t)Bn * Lenc * FF); // 8192*2048
    float* enc_out = alloc((size_t)Bn * 512 * D);
    float* stats   = alloc(1024);
    float* decx    = encx;   // decoder reuses encoder activation buffer
    float* troll   = tq;     // conv-layer scratch (attn temps free by then)
    float* convy   = tk;

    _Float16* hbase = (_Float16*)(ws + off);
    size_t hoff = 0;
    auto halloc = [&](size_t n) { _Float16* p = hbase + hoff; hoff += n; return p; };
    auto prep = [&](const float* W, int K, int N) {   // -> f16 transposed [N][K]
        _Float16* p = halloc((size_t)K * N);
        const int total = K * N;
        convert_transpose_kernel<<<(total + 255) / 256, 256, 0, stream>>>(W, p, K, N);
        return p;
    };

    // pre-convert + transpose all GEMM weights (static per launch)
    _Float16* e0_wq = prep(IN(48), D, D); _Float16* e0_wk = prep(IN(46), D, D);
    _Float16* e0_wv = prep(IN(49), D, D); _Float16* e0_wo = prep(IN(47), D, D);
    _Float16* e0_w1 = prep(IN(56), D, FF); _Float16* e0_w2 = prep(IN(57), FF, D);
    _Float16* e1_wq = prep(IN(64), D, D); _Float16* e1_wk = prep(IN(62), D, D);
    _Float16* e1_wv = prep(IN(65), D, D); _Float16* e1_wo = prep(IN(63), D, D);
    _Float16* e1_w1 = prep(IN(72), D, FF); _Float16* e1_w2 = prep(IN(73), FF, D);
    _Float16* cv_w0 = prep(IN(38) + 0 * D * D, D, D);
    _Float16* cv_w1 = prep(IN(38) + 1 * D * D, D, D);
    _Float16* cv_w2 = prep(IN(38) + 2 * D * D, D, D);
    _Float16* ds_wq = prep(IN(29), D, D); _Float16* ds_wk = prep(IN(27), D, D);
    _Float16* ds_wv = prep(IN(30), D, D); _Float16* ds_wo = prep(IN(28), D, D);
    _Float16* dc_wq = prep(IN(15), D, D); _Float16* dc_wk = prep(IN(13), D, D);
    _Float16* dc_wv = prep(IN(16), D, D); _Float16* dc_wo = prep(IN(14), D, D);
    _Float16* d_w1  = prep(IN(31), D, FF); _Float16* d_w2 = prep(IN(32), FF, D);

    auto gemm = [&](const float* A, const _Float16* Bt, const float* bias, float* C,
                    int M, int N, int K, int flags) {
        dim3 grid(N / GEMM_TN, (M + GEMM_TM - 1) / GEMM_TM);
        gemm_wmma_kernel<<<grid, 256, 0, stream>>>(A, Bt, bias, C, M, N, K, flags);
    };
    auto ln = [&](const float* X, const float* R, const float* g, const float* b,
                  float* O, int rows) {
        ln_kernel<<<(rows + 7) / 8, 256, 0, stream>>>(X, R, g, b, O, rows, D);
    };
    auto attn_block = [&](const float* xq, const float* xkv, int Mq, int Mkv,
                          int Lq, int Lk,
                          const _Float16* wq, const float* bq, const _Float16* wk,
                          const float* bk, const _Float16* wv, const float* bv,
                          const _Float16* wo, const float* bo,
                          float* outp, int causal) {
        gemm(xq,  wq, bq, tq, Mq,  D, D, 1);
        gemm(xkv, wk, bk, tk, Mkv, D, D, 1);
        gemm(xkv, wv, bv, tv, Mkv, D, D, 1);
        dim3 g(Lq / 8, 8, Bn);
        attention_kernel<<<g, 256, 0, stream>>>(tq, tk, tv, tctx, Lq, Lk, D, 64, causal);
        gemm(tctx, wo, bo, outp, Mq, D, D, 1);
    };
    auto enc_layer = [&](float* X, int Lcur,
                         const _Float16* wq, const float* bq, const _Float16* wk,
                         const float* bk, const _Float16* wv, const float* bv,
                         const _Float16* wo, const float* bo,
                         const _Float16* w1, const float* b1,
                         const _Float16* w2, const float* b2,
                         const float* n1g, const float* n1b,
                         const float* n2g, const float* n2b) {
        const int M = Bn * Lcur;
        attn_block(X, X, M, M, Lcur, Lcur, wq, bq, wk, bk, wv, bv, wo, bo, tmpa, 0);
        ln(X, tmpa, n1g, n1b, x2, M);
        gemm(x2,  w1, b1, ffh,  M, FF, D, 1 | 2);   // bias + GELU
        gemm(ffh, w2, b2, tmpa, M, D, FF, 1);
        ln(x2, tmpa, n2g, n2b, X, M);
    };

    const int thr = 256;
    const int totE = Bn * Lenc * D;

    // ================= encoder =================
    embed_kernel<<<(totE + thr - 1) / thr, thr, 0, stream>>>(
        x_enc, x_mark_enc, IN(39), IN(41), IN(40), encx, Bn, Lenc, 7, D, 4);

    enc_layer(encx, Lenc, e0_wq, IN(44), e0_wk, IN(42), e0_wv, IN(45), e0_wo, IN(43),
              e0_w1, IN(50), e0_w2, IN(51), IN(53), IN(52), IN(55), IN(54));

    // distillation conv: y = roll(x,+1)w0 + x w1 + roll(x,-1)w2 + b ; BN ; ELU ; pool
    {
        const int M = Bn * Lenc;
        roll_rows_kernel<<<(totE + thr - 1) / thr, thr, 0, stream>>>(encx, troll, Bn, Lenc, D, -1);
        gemm(troll, cv_w0, nullptr, convy, M, D, D, 0);
        gemm(encx,  cv_w1, nullptr, convy, M, D, D, 4);
        roll_rows_kernel<<<(totE + thr - 1) / thr, thr, 0, stream>>>(encx, troll, Bn, Lenc, D, +1);
        gemm(troll, cv_w2, IN(35), convy, M, D, D, 4 | 1);
        chan_stats_kernel<<<D, 256, 0, stream>>>(convy, stats, stats + D, M, D);
        bn_elu_kernel<<<(totE + thr - 1) / thr, thr, 0, stream>>>(convy, stats, stats + D,
                                                                  IN(37), IN(36), totE, D);
        maxpool_kernel<<<(totE / 2 + thr - 1) / thr, thr, 0, stream>>>(convy, encx, Bn, Lenc, D);
    }

    enc_layer(encx, 512, e1_wq, IN(60), e1_wk, IN(58), e1_wv, IN(61), e1_wo, IN(59),
              e1_w1, IN(66), e1_w2, IN(67), IN(69), IN(68), IN(71), IN(70));

    ln(encx, nullptr, IN(75), IN(74), enc_out, Bn * 512);

    // ================= decoder =================
    const int totD = Bn * Ldec * D;
    const int Md = Bn * Ldec;
    embed_kernel<<<(totD + thr - 1) / thr, thr, 0, stream>>>(
        x_dec, x_mark_dec, IN(4), IN(6), IN(5), decx, Bn, Ldec, 7, D, 4);

    // self attention (causal)
    attn_block(decx, decx, Md, Md, Ldec, Ldec,
               ds_wq, IN(25), ds_wk, IN(23), ds_wv, IN(26), ds_wo, IN(24), tmpa, 1);
    ln(decx, tmpa, IN(18), IN(17), x2, Md);
    // cross attention vs encoder output (Lk = 512)
    attn_block(x2, enc_out, Md, Bn * 512, Ldec, 512,
               dc_wq, IN(11), dc_wk, IN(9), dc_wv, IN(12), dc_wo, IN(10), tmpa, 0);
    ln(x2, tmpa, IN(20), IN(19), decx, Md);
    // FFN
    gemm(decx, d_w1, IN(7), ffh,  Md, FF, D, 1 | 2);
    gemm(ffh,  d_w2, IN(8), tmpa, Md, D, FF, 1);
    ln(decx, tmpa, IN(22), IN(21), x2, Md);
    // final decoder norm
    ln(x2, nullptr, IN(34), IN(33), decx, Md);

    // projection of last PRED steps -> [B, PRED, 7]
    const int totO = Bn * PRED * 7;
    proj_kernel<<<(totO + thr - 1) / thr, thr, 0, stream>>>(
        decx, IN(77), IN(76), (float*)d_out, Bn, Ldec, PRED, D, 7);
#undef IN
}
